// InteractionGNNCell_23974507446588
// MI455X (gfx1250) — matrix operand order repaired and checked by reading, not compiled, not run on hardware
//
#include <hip/hip_runtime.h>
#include <hip/hip_bf16.h>

// ---------------------------------------------------------------------------
// Types for CDNA5 WMMA bf16: D(f32 16x16) = A(bf16 16x32) * B(bf16 32x16) + C
// ---------------------------------------------------------------------------
typedef __attribute__((ext_vector_type(16))) __bf16 v16bf;
typedef __attribute__((ext_vector_type(8)))  __bf16 v8bf;
typedef __attribute__((ext_vector_type(8)))  float  v8f;

__device__ __forceinline__ unsigned short f2bf_bits(float f) {
    union { float f; unsigned int u; } c; c.f = f;
    unsigned int u = c.u;
    // round-to-nearest-even bf16
    return (unsigned short)((u + 0x7FFFu + ((u >> 16) & 1u)) >> 16);
}

__device__ __forceinline__ float gelu_exact(float x) {
    return 0.5f * x * (1.0f + erff(x * 0.70710678118654752f));
}

// B fragment: 32 contiguous bytes per lane (packed layout)
__device__ __forceinline__ v16bf load_fragB(const unsigned short* p) {
    v8bf lo = *(const v8bf*)(p);
    v8bf hi = *(const v8bf*)(p + 8);
    return __builtin_shufflevector(lo, hi,
        0, 1, 2, 3, 4, 5, 6, 7, 8, 9, 10, 11, 12, 13, 14, 15);
}

// A fragment: two 16B chunks 32B apart (ISA 16-bit A 16x32 K-interleave)
__device__ __forceinline__ v16bf load_fragA(const unsigned short* p) {
    v8bf lo = *(const v8bf*)(p);
    v8bf hi = *(const v8bf*)(p + 16);
    return __builtin_shufflevector(lo, hi,
        0, 1, 2, 3, 4, 5, 6, 7, 8, 9, 10, 11, 12, 13, 14, 15);
}

__device__ __forceinline__ v8f wmma_bf16(v16bf a, v16bf b, v8f c) {
    return __builtin_amdgcn_wmma_f32_16x16x32_bf16(
        false, a, false, b, (short)0, c, false, false);
}

// ---------------------------------------------------------------------------
// Pack fp32 weight W[K x Nout] (row-major, x @ W convention) into bf16 WMMA
// B-fragment layout. Per 32x16 (KxN) tile: lane l holds column (l&15);
// lanes 0-15 hold K=tk*32+{0..7,16..23}, lanes 16-31 hold K=tk*32+{8..15,24..31}, 16 contiguous
// bf16 values per lane => vectorizable 32B fragment loads in the GEMM.
// Wp index = ((tn * (K/32) + tk) * 512) + lane*16 + i
// ---------------------------------------------------------------------------
__global__ void pack_w_kernel(const float* __restrict__ W,
                              unsigned short* __restrict__ Wp,
                              int K, int Nout) {
    int idx = blockIdx.x * blockDim.x + threadIdx.x;
    int total = K * Nout;
    if (idx >= total) return;
    int frag   = idx >> 9;        // /512
    int within = idx & 511;
    int lane   = within >> 4;
    int i      = within & 15;
    int kfrag  = K >> 5;
    int tk = frag % kfrag;
    int tn = frag / kfrag;
    int col = tn * 16 + (lane & 15);
    int kk  = tk * 32 + (lane >> 4) * 8 + (i & 7) + (i >> 3) * 16;
    Wp[idx] = f2bf_bits(W[(size_t)kk * Nout + col]);
}

// ---------------------------------------------------------------------------
// Scatter-add edge features into per-node message buffer (segment_sum by dst)
// ---------------------------------------------------------------------------
__global__ void scatter_add_kernel(const float* __restrict__ edges,
                                   const int* __restrict__ dstI,
                                   float* __restrict__ msg, int E) {
    size_t idx = (size_t)blockIdx.x * blockDim.x + threadIdx.x;
    size_t total = (size_t)E * 128;
    if (idx >= total) return;
    int    c = (int)(idx & 127);
    size_t e = idx >> 7;
    atomicAdd(&msg[(size_t)dstI[e] * 128 + c], edges[idx]);
}

// ---------------------------------------------------------------------------
// Per-column batch stats for node input concat([nodes, msg]) -> scale/shift
// scale = w * rsqrt(var + eps), shift = b - mean * scale
// ---------------------------------------------------------------------------
__global__ void node_stats_kernel(const float* __restrict__ nodes,
                                  const float* __restrict__ msg,
                                  const float* __restrict__ w,
                                  const float* __restrict__ b,
                                  float* __restrict__ scale,
                                  float* __restrict__ shift, int Nrows) {
    int col = blockIdx.x;  // 0..255
    const float* base = (col < 128) ? (nodes + col) : (msg + (col - 128));
    float s = 0.f, s2 = 0.f;
    for (int r = threadIdx.x; r < Nrows; r += blockDim.x) {
        float v = base[(size_t)r * 128];
        s += v; s2 += v * v;
    }
    __shared__ float sh[256], sh2[256];
    sh[threadIdx.x] = s; sh2[threadIdx.x] = s2;
    __syncthreads();
    for (int off = 128; off > 0; off >>= 1) {
        if (threadIdx.x < off) {
            sh[threadIdx.x]  += sh[threadIdx.x + off];
            sh2[threadIdx.x] += sh2[threadIdx.x + off];
        }
        __syncthreads();
    }
    if (threadIdx.x == 0) {
        float inv = 1.0f / (float)Nrows;
        float mean = sh[0] * inv;
        float var  = sh2[0] * inv - mean * mean;
        float sc = w[col] * rsqrtf(var + 1e-5f);
        scale[col] = sc;
        shift[col] = b[col] - mean * sc;
    }
}

// Per-column stats for edge input concat([nodesOut[src], nodesOut[dst], edges])
__global__ void edge_stats_kernel(const float* __restrict__ nodesOut,
                                  const float* __restrict__ edges,
                                  const int* __restrict__ srcI,
                                  const int* __restrict__ dstI,
                                  const float* __restrict__ w,
                                  const float* __restrict__ b,
                                  float* __restrict__ scale,
                                  float* __restrict__ shift, int E) {
    int col = blockIdx.x;  // 0..383
    float s = 0.f, s2 = 0.f;
    for (int r = threadIdx.x; r < E; r += blockDim.x) {
        float v;
        if (col < 128)      v = nodesOut[(size_t)srcI[r] * 128 + col];
        else if (col < 256) v = nodesOut[(size_t)dstI[r] * 128 + (col - 128)];
        else                v = edges[(size_t)r * 128 + (col - 256)];
        s += v; s2 += v * v;
    }
    __shared__ float sh[256], sh2[256];
    sh[threadIdx.x] = s; sh2[threadIdx.x] = s2;
    __syncthreads();
    for (int off = 128; off > 0; off >>= 1) {
        if (threadIdx.x < off) {
            sh[threadIdx.x]  += sh[threadIdx.x + off];
            sh2[threadIdx.x] += sh2[threadIdx.x + off];
        }
        __syncthreads();
    }
    if (threadIdx.x == 0) {
        float inv = 1.0f / (float)E;
        float mean = sh[0] * inv;
        float var  = sh2[0] * inv - mean * mean;
        float sc = w[col] * rsqrtf(var + 1e-5f);
        scale[col] = sc;
        shift[col] = b[col] - mean * sc;
    }
}

// ---------------------------------------------------------------------------
// Materialize batch-normalized inputs as bf16 row-major activations
// ---------------------------------------------------------------------------
__global__ void bn_node_kernel(const float* __restrict__ nodes,
                               const float* __restrict__ msg,
                               const float* __restrict__ scale,
                               const float* __restrict__ shift,
                               unsigned short* __restrict__ Xn, int Nrows) {
    size_t idx = (size_t)blockIdx.x * blockDim.x + threadIdx.x;
    size_t total = (size_t)Nrows * 256;
    if (idx >= total) return;
    int    col = (int)(idx & 255);
    size_t row = idx >> 8;
    float v = (col < 128) ? nodes[row * 128 + col] : msg[row * 128 + (col - 128)];
    Xn[idx] = f2bf_bits(v * scale[col] + shift[col]);
}

__global__ void bn_edge_kernel(const float* __restrict__ nodesOut,
                               const float* __restrict__ edges,
                               const int* __restrict__ srcI,
                               const int* __restrict__ dstI,
                               const float* __restrict__ scale,
                               const float* __restrict__ shift,
                               unsigned short* __restrict__ Xe,
                               int rowBase, int rowCount) {
    size_t idx = (size_t)blockIdx.x * blockDim.x + threadIdx.x;
    size_t total = (size_t)rowCount * 384;
    if (idx >= total) return;
    int    col = (int)(idx % 384);
    size_t lr  = idx / 384;
    size_t r   = (size_t)rowBase + lr;
    float v;
    if (col < 128)      v = nodesOut[(size_t)srcI[r] * 128 + col];
    else if (col < 256) v = nodesOut[(size_t)dstI[r] * 128 + (col - 128)];
    else                v = edges[r * 128 + (col - 256)];
    Xe[lr * 384 + col] = f2bf_bits(v * scale[col] + shift[col]);
}

// ---------------------------------------------------------------------------
// WMMA bf16 GEMM:  Y[M x Nout] = act( X[M x K](bf16) @ Wp(bf16 packed) + bias )
// mode 0: GELU(exact) epilogue, store bf16 to Ybf
// mode 1: + residual[M x Nout] epilogue, store fp32 to Yf
//
// Each wave owns a 32-row strip (two 16-row m-tiles that SHARE every B
// fragment -> 1.5 b128 loads per v_wmma) x 64-col n-block (8 accumulators).
// K-steps run in pairs with two disjoint fragment register sets: while the
// matrix pipe consumes set 0 the loads fill set 1 and vice versa.
// Addressing uses LOOP-CARRIED POINTERS with constant strides: half-step
// loads use immediate offsets (folded into global_load), and per pair only 6
// pointer bumps execute -- these write pointer registers, never fragment
// registers, so no WMMA->VALU WAR hazard (no v_nop padding). The final pair
// uses a scalar-selected step of 0 (branch-free, no OOB reads).
// kfrag is always even (K in {256,384,512}).
// __launch_bounds__(256, 1) lifts the VGPR budget (no scratch spills).
// X must be padded to a multiple of 32 rows (stores are guarded by orow < M).
// All branches around WMMAs are wave-uniform => EXEC all-ones (ISA req).
// ---------------------------------------------------------------------------
__global__ void __launch_bounds__(256, 1)
wmma_gemm_bf16_kernel(const unsigned short* __restrict__ X,
                      const unsigned short* __restrict__ Wp,
                      const float* __restrict__ bias,
                      const float* __restrict__ residual,
                      float* __restrict__ Yf,
                      unsigned short* __restrict__ Ybf,
                      int M, int K, int Nout, int mode) {
    const int WAVES = 8;
    int wave = threadIdx.x >> 5;
    int lane = threadIdx.x & 31;
    int Mstrips = (M + 31) >> 5;            // 32 rows per wave
    int mstrip  = blockIdx.x * WAVES + wave;
    if (mstrip >= Mstrips) return;

    int m0    = mstrip << 5;                // rows m0 .. m0+31
    int hi    = lane >> 4;                  // half-wave selector
    int nlane = lane & 15;

    // A-fragment row bases for the two m-tiles
    const unsigned short* Xr0 = X + (size_t)(m0 + nlane) * K + hi * 8;
    const unsigned short* Xr1 = Xr0 + (size_t)16 * K;
    int kfrag = K >> 5;
    size_t tstride = (size_t)kfrag << 9;    // B elements per n-tile column

    for (int n0 = 0; n0 < Nout; n0 += 64) {
        v8f acc0[4], acc1[4];
#pragma unroll
        for (int t = 0; t < 4; ++t) {
            float bv = bias[n0 + t * 16 + nlane];
#pragma unroll
            for (int r = 0; r < 8; ++r) { acc0[t][r] = bv; acc1[t][r] = bv; }
        }

        // Loop-carried fragment pointers (bumped by constant strides).
        const unsigned short* pA0 = Xr0;
        const unsigned short* pA1 = Xr1;
        const unsigned short* pB0 =
            Wp + ((size_t)(n0 >> 4) * tstride) + lane * 16;
        const unsigned short* pB1 = pB0 + tstride;
        const unsigned short* pB2 = pB1 + tstride;
        const unsigned short* pB3 = pB2 + tstride;

        // ---- double-buffered K pipeline (kfrag is even) ----
        v16bf a00 = load_fragA(pA0);
        v16bf a01 = load_fragA(pA1);
        v16bf b00 = load_fragB(pB0);
        v16bf b01 = load_fragB(pB1);
        v16bf b02 = load_fragB(pB2);
        v16bf b03 = load_fragB(pB3);

        for (int kt = 0; kt < kfrag; kt += 2) {
            // fill set 1 (kt+1) via immediate offsets while set 0 computes
            v16bf a10 = load_fragA(pA0 + 32);
            v16bf a11 = load_fragA(pA1 + 32);
            v16bf b10 = load_fragB(pB0 + 512);
            v16bf b11 = load_fragB(pB1 + 512);
            v16bf b12 = load_fragB(pB2 + 512);
            v16bf b13 = load_fragB(pB3 + 512);

            acc0[0] = wmma_bf16(a00, b00, acc0[0]);
            acc1[0] = wmma_bf16(a01, b00, acc1[0]);
            acc0[1] = wmma_bf16(a00, b01, acc0[1]);
            acc1[1] = wmma_bf16(a01, b01, acc1[1]);
            acc0[2] = wmma_bf16(a00, b02, acc0[2]);
            acc1[2] = wmma_bf16(a01, b02, acc1[2]);
            acc0[3] = wmma_bf16(a00, b03, acc0[3]);
            acc1[3] = wmma_bf16(a01, b03, acc1[3]);

            // advance pointers (scalar-selected 0 step on the final pair so
            // the refill harmlessly reloads consumed data -- branch-free)
            int stepA = (kt + 2 < kfrag) ? 64 : 0;
            int stepB = (kt + 2 < kfrag) ? 1024 : 0;
            pA0 += stepA; pA1 += stepA;
            pB0 += stepB; pB1 += stepB; pB2 += stepB; pB3 += stepB;

            // refill set 0 (kt+2) while set 1 computes
            a00 = load_fragA(pA0);
            a01 = load_fragA(pA1);
            b00 = load_fragB(pB0);
            b01 = load_fragB(pB1);
            b02 = load_fragB(pB2);
            b03 = load_fragB(pB3);

            acc0[0] = wmma_bf16(a10, b10, acc0[0]);
            acc1[0] = wmma_bf16(a11, b10, acc1[0]);
            acc0[1] = wmma_bf16(a10, b11, acc0[1]);
            acc1[1] = wmma_bf16(a11, b11, acc1[1]);
            acc0[2] = wmma_bf16(a10, b12, acc0[2]);
            acc1[2] = wmma_bf16(a11, b12, acc1[2]);
            acc0[3] = wmma_bf16(a10, b13, acc0[3]);
            acc1[3] = wmma_bf16(a11, b13, acc1[3]);
        }

        // Epilogue. C/D layout: VGPR r, lanes 0-15 -> row r, col=lane;
        // lanes 16-31 -> row 8+r, col=lane-16. Stores guarded for the
        // (single) partial strip when M % 32 == 16.
#pragma unroll
        for (int mt = 0; mt < 2; ++mt) {
            int mb = m0 + mt * 16;
#pragma unroll
            for (int t = 0; t < 4; ++t) {
                int col = n0 + t * 16 + nlane;
#pragma unroll
                for (int r = 0; r < 8; ++r) {
                    int orow = mb + r + hi * 8;
                    float v = (mt == 0) ? acc0[t][r] : acc1[t][r];
                    if (orow < M) {
                        if (mode == 0) {
                            Ybf[(size_t)orow * Nout + col] =
                                f2bf_bits(gelu_exact(v));
                        } else {
                            Yf[(size_t)orow * Nout + col] =
                                v + residual[(size_t)orow * Nout + col];
                        }
                    }
                }
            }
        }
    }
}

// ---------------------------------------------------------------------------
// Host-side orchestration (graph-capture safe: kernels + hipMemsetAsync only)
// ---------------------------------------------------------------------------
extern "C" void kernel_launch(void* const* d_in, const int* in_sizes, int n_in,
                              void* d_out, int out_size, void* d_ws, size_t ws_size,
                              hipStream_t stream) {
    const float* nodes = (const float*)d_in[0];
    const float* edges = (const float*)d_in[1];
    const int*   graph = (const int*)d_in[2];
    const float* nnw   = (const float*)d_in[3];
    const float* nnb   = (const float*)d_in[4];
    const float* enw   = (const float*)d_in[5];
    const float* enb   = (const float*)d_in[6];
    const float* Wn1   = (const float*)d_in[7];
    const float* bn1   = (const float*)d_in[8];
    const float* Wn2   = (const float*)d_in[9];
    const float* bn2   = (const float*)d_in[10];
    const float* We1   = (const float*)d_in[11];
    const float* be1   = (const float*)d_in[12];
    const float* We2   = (const float*)d_in[13];
    const float* be2   = (const float*)d_in[14];

    const int N  = in_sizes[0] / 128;   // 50000
    const int E  = in_sizes[1] / 128;   // 320000
    const int NP = (N + 31) & ~31;      // padded rows for 32-row wave strips
    const int* srcI = graph;
    const int* dstI = graph + E;

    float* outNodes = (float*)d_out;
    float* outEdges = outNodes + (size_t)N * 128;

    // Workspace carve-out (all regions 256B aligned)
    char* ws = (char*)d_ws;
    size_t off = 0;
    auto carve = [&](size_t bytes) -> void* {
        void* p = ws + off;
        off += (bytes + 255) & ~(size_t)255;
        return p;
    };
    float*          msg    = (float*)carve((size_t)N * 128 * 4);
    float*          scaleN = (float*)carve(256 * 4);
    float*          shiftN = (float*)carve(256 * 4);
    float*          scaleE = (float*)carve(384 * 4);
    float*          shiftE = (float*)carve(384 * 4);
    unsigned short* Wn1p   = (unsigned short*)carve((size_t)256 * 512 * 2);
    unsigned short* Wn2p   = (unsigned short*)carve((size_t)512 * 128 * 2);
    unsigned short* We1p   = (unsigned short*)carve((size_t)384 * 512 * 2);
    unsigned short* We2p   = (unsigned short*)carve((size_t)512 * 128 * 2);
    unsigned short* Xn     = (unsigned short*)carve((size_t)NP * 256 * 2);
    unsigned short* H1     = (unsigned short*)carve((size_t)NP * 512 * 2);
    const int CH = 40000;               // edge-row chunk (multiple of 32)
    unsigned short* Xe     = (unsigned short*)carve((size_t)CH * 384 * 2);
    unsigned short* H2     = (unsigned short*)carve((size_t)CH * 512 * 2);

    hipMemsetAsync(msg, 0, (size_t)N * 128 * 4, stream);

    // Pack weights into WMMA fragment layout (bf16)
    pack_w_kernel<<<(256 * 512 + 255) / 256, 256, 0, stream>>>(Wn1, Wn1p, 256, 512);
    pack_w_kernel<<<(512 * 128 + 255) / 256, 256, 0, stream>>>(Wn2, Wn2p, 512, 128);
    pack_w_kernel<<<(384 * 512 + 255) / 256, 256, 0, stream>>>(We1, We1p, 384, 512);
    pack_w_kernel<<<(512 * 128 + 255) / 256, 256, 0, stream>>>(We2, We2p, 512, 128);

    // segment_sum(edges, dst)
    {
        size_t total = (size_t)E * 128;
        unsigned blocks = (unsigned)((total + 255) / 256);
        scatter_add_kernel<<<blocks, 256, 0, stream>>>(edges, dstI, msg, E);
    }

    // ---- node path ----
    node_stats_kernel<<<256, 256, 0, stream>>>(nodes, msg, nnw, nnb, scaleN, shiftN, N);
    {
        size_t total = (size_t)N * 256;
        bn_node_kernel<<<(unsigned)((total + 255) / 256), 256, 0, stream>>>(
            nodes, msg, scaleN, shiftN, Xn, N);
    }
    int strN = (NP / 32 + 7) / 8;       // blocks of 8 waves x 32 rows
    wmma_gemm_bf16_kernel<<<strN, 256, 0, stream>>>(
        Xn, Wn1p, bn1, nullptr, nullptr, H1, N, 256, 512, 0);
    wmma_gemm_bf16_kernel<<<strN, 256, 0, stream>>>(
        H1, Wn2p, bn2, nodes, outNodes, nullptr, N, 512, 128, 1);

    // ---- edge path ----
    edge_stats_kernel<<<384, 256, 0, stream>>>(outNodes, edges, srcI, dstI,
                                               enw, enb, scaleE, shiftE, E);
    for (int base = 0; base < E; base += CH) {
        int rows = (E - base) < CH ? (E - base) : CH;
        size_t total = (size_t)rows * 384;
        bn_edge_kernel<<<(unsigned)((total + 255) / 256), 256, 0, stream>>>(
            outNodes, edges, srcI, dstI, scaleE, shiftE, Xe, base, rows);
        int strE = ((rows + 31) / 32 + 7) / 8;
        wmma_gemm_bf16_kernel<<<strE, 256, 0, stream>>>(
            Xe, We1p, be1, nullptr, nullptr, H2, rows, 384, 512, 0);
        wmma_gemm_bf16_kernel<<<strE, 256, 0, stream>>>(
            H2, We2p, be2, edges + (size_t)base * 128,
            outEdges + (size_t)base * 128, nullptr, rows, 512, 128, 1);
    }
}
